// NonLocalConvBlock_72859825209893
// MI455X (gfx1250) — compile-verified, matrix operand order checked by reading, not run on hardware
//
#include <hip/hip_runtime.h>
#include <hip/hip_bf16.h>

typedef __attribute__((ext_vector_type(16))) _Float16 v16h;
typedef __attribute__((ext_vector_type(8)))  _Float16 v8h;
typedef __attribute__((ext_vector_type(8)))  float    v8f;

#define N_TOK 9216
#define C_DIM 128
#define CI    64

// Load a 16x32 f16 A/B WMMA fragment from a row-major matrix.
// CDNA5 layout (ISA 7.12.2): lane L (L<16): row=L, K in {0..7,16..23};
// lane L>=16: row=L-16, K in {8..15,24..31}. Each 8-half group is
// contiguous -> two global_load_b128 per fragment.
__device__ __forceinline__ v16h load_frag16(const _Float16* __restrict__ p,
                                            int row, int ld, int kbase, int hb) {
  const _Float16* q = p + (long)row * ld + kbase + hb * 8;
  v8h lo = *(const v8h*)q;
  v8h hi = *(const v8h*)(q + 16);
  v16h r;
#pragma unroll
  for (int i = 0; i < 8; ++i) { r[i] = lo[i]; r[8 + i] = hi[i]; }
  return r;
}

__device__ __forceinline__ v8f wmma_f16(v16h a, v16h b, v8f c) {
  return __builtin_amdgcn_wmma_f32_16x16x32_f16(false, a, false, b, (short)0, c,
                                                false, false);
}

// Swap data between the two 16-lane halves (lane <-> lane^16).
__device__ __forceinline__ float xor16(float v) {
#if __has_builtin(__builtin_amdgcn_permlanex16)
  int i = __builtin_bit_cast(int, v);
  int r = __builtin_amdgcn_permlanex16(i, i, 0x76543210, 0xfedcba98, false, false);
  return __builtin_bit_cast(float, r);
#else
  return __shfl_xor(v, 16);
#endif
}

// ---------------- Kernel 1: projections (theta/phi row-major, g transposed) --
__global__ __launch_bounds__(256) void proj_kernel(
    const float* __restrict__ x,
    const float* __restrict__ tw, const float* __restrict__ tb,
    const float* __restrict__ pw, const float* __restrict__ pb,
    const float* __restrict__ gw, const float* __restrict__ gb,
    const float* __restrict__ ow,
    _Float16* __restrict__ theta, _Float16* __restrict__ phi,
    _Float16* __restrict__ gT, _Float16* __restrict__ owh) {
  int n = blockIdx.x * blockDim.x + threadIdx.x;  // token id
  if (n < C_DIM * CI) owh[n] = (_Float16)ow[n];   // out_w -> f16 (128x64 row-major)
  if (n >= N_TOK) return;

  for (int proj = 0; proj < 3; ++proj) {
    const float* w = (proj == 0) ? tw : (proj == 1) ? pw : gw;
    const float* b = (proj == 0) ? tb : (proj == 1) ? pb : gb;
#pragma unroll
    for (int ch = 0; ch < 4; ++ch) {
      float acc[16];
#pragma unroll
      for (int i = 0; i < 16; ++i) acc[i] = b[ch * 16 + i];
      for (int c = 0; c < C_DIM; ++c) {
        float xv = x[(long)c * N_TOK + n];  // coalesced across threads
        const float* wr = w + (ch * 16) * C_DIM + c;
#pragma unroll
        for (int i = 0; i < 16; ++i)
          acc[i] = __builtin_fmaf(xv, wr[i * C_DIM], acc[i]);
      }
      if (proj == 0) {
#pragma unroll
        for (int i = 0; i < 16; ++i)
          theta[(long)n * CI + ch * 16 + i] = (_Float16)acc[i];
      } else if (proj == 1) {
#pragma unroll
        for (int i = 0; i < 16; ++i)
          phi[(long)n * CI + ch * 16 + i] = (_Float16)acc[i];
      } else {
#pragma unroll
        for (int i = 0; i < 16; ++i)
          gT[(long)(ch * 16 + i) * N_TOK + n] = (_Float16)acc[i];
      }
    }
  }
}

// ------------- Kernel 2: transposed flash attention, software-pipelined ------
// S^T = phi @ theta^T: queries on lanes, keys on VGPRs. P^T in D-layout IS the
// B-fragment layout of the PV WMMA; y^T accumulators ARE the B-fragments of the
// output projection. Key-chunk loads are double-buffered (ping-pong) so one
// chunk's global_load_b128s overlap the previous chunk's WMMAs + softmax.
__global__ __launch_bounds__(256) void flash_kernel(
    const _Float16* __restrict__ theta, const _Float16* __restrict__ phi,
    const _Float16* __restrict__ gT, const _Float16* __restrict__ owh,
    const float* __restrict__ out_b, const float* __restrict__ x,
    float* __restrict__ out) {
  const int lane = threadIdx.x & 31;
  const int wave = threadIdx.x >> 5;
  const int hb = lane >> 4;   // lane half (K-group / M+8 select)
  const int lx = lane & 15;   // query column within tile
  const int qbase = (blockIdx.x * 8 + wave) * 16;  // 16 queries per wave

  // theta^T as resident B-fragments: B[k][q] = theta[q][k]
  const v16h bt0 = load_frag16(theta, qbase + lx, CI, 0, hb);
  const v16h bt1 = load_frag16(theta, qbase + lx, CI, 32, hb);

  // O^T accumulators: 64 channels x 16 queries (channels on VGPRs, q on lanes)
  v8f acc0 = {}, acc1 = {}, acc2 = {}, acc3 = {};
  float m = -1e30f, l = 0.0f;  // per-query (per-lane) online-softmax stats

  auto load_chunk = [&](int kb, v16h& p00, v16h& p01, v16h& p10, v16h& p11,
                        v16h& g0, v16h& g1, v16h& g2, v16h& g3) {
    p00 = load_frag16(phi, kb + lx, CI, 0, hb);
    p01 = load_frag16(phi, kb + lx, CI, 32, hb);
    p10 = load_frag16(phi, kb + 16 + lx, CI, 0, hb);
    p11 = load_frag16(phi, kb + 16 + lx, CI, 32, hb);
    g0 = load_frag16(gT, 0 * 16 + lx, N_TOK, kb, hb);
    g1 = load_frag16(gT, 1 * 16 + lx, N_TOK, kb, hb);
    g2 = load_frag16(gT, 2 * 16 + lx, N_TOK, kb, hb);
    g3 = load_frag16(gT, 3 * 16 + lx, N_TOK, kb, hb);
  };

  auto process = [&](v16h p00, v16h p01, v16h p10, v16h p11,
                     v16h g0, v16h g1, v16h g2, v16h g3) {
    // S^T tiles: s0 = first 16 keys, s1 = next 16 (per lane: 8 each + partner)
    v8f s0 = {}, s1 = {};
    s0 = wmma_f16(p00, bt0, s0);
    s0 = wmma_f16(p01, bt1, s0);
    s1 = wmma_f16(p10, bt0, s1);
    s1 = wmma_f16(p11, bt1, s1);

    // per-query online softmax: in-register reduce + one half-swap
    float mx = -1e30f;
#pragma unroll
    for (int r = 0; r < 8; ++r) mx = fmaxf(mx, fmaxf(s0[r], s1[r]));
    mx = fmaxf(mx, xor16(mx));
    float mnew = fmaxf(m, mx);
    float scale = __expf(m - mnew);
    m = mnew;
    float rs = 0.0f;
#pragma unroll
    for (int r = 0; r < 8; ++r) {
      s0[r] = __expf(s0[r] - mnew);
      s1[r] = __expf(s1[r] - mnew);
      rs += s0[r] + s1[r];
    }
    rs += xor16(rs);
    l = l * scale + rs;
#pragma unroll
    for (int r = 0; r < 8; ++r) {  // rescale O^T by per-lane scalar
      acc0[r] *= scale; acc1[r] *= scale; acc2[r] *= scale; acc3[r] *= scale;
    }

    // P^T in D-layout == B-fragment layout: pure in-register f32->f16 pack
    v16h pb;
#pragma unroll
    for (int r = 0; r < 8; ++r) {
      pb[r]     = (_Float16)s0[r];   // K = hb*8 + r
      pb[8 + r] = (_Float16)s1[r];   // K = 16 + hb*8 + r
    }

    // O^T += g^T_tile x P^T   (K = 32 keys)
    acc0 = wmma_f16(g0, pb, acc0);
    acc1 = wmma_f16(g1, pb, acc1);
    acc2 = wmma_f16(g2, pb, acc2);
    acc3 = wmma_f16(g3, pb, acc3);
  };

  // ping-pong fragment buffers (keeps loads a full chunk ahead of their use)
  v16h A0, A1, A2, A3, A4, A5, A6, A7;
  v16h B0, B1, B2, B3, B4, B5, B6, B7;
  load_chunk(0, A0, A1, A2, A3, A4, A5, A6, A7);

  for (int kb = 0; kb < N_TOK; kb += 64) {  // 2 chunks (64 keys) per iteration
    load_chunk(kb + 32, B0, B1, B2, B3, B4, B5, B6, B7);
    process(A0, A1, A2, A3, A4, A5, A6, A7);
    const int nkb = (kb + 64 < N_TOK) ? kb + 64 : 0;  // last load is a dummy
    load_chunk(nkb, A0, A1, A2, A3, A4, A5, A6, A7);
    process(B0, B1, B2, B3, B4, B5, B6, B7);
  }

  // --- finalize: y = O/l, softmax over 64 channels (channels on VGPRs) ---
  float inv = 1.0f / l;
  float cmx = -1e30f;
#pragma unroll
  for (int r = 0; r < 8; ++r) {
    acc0[r] *= inv; acc1[r] *= inv; acc2[r] *= inv; acc3[r] *= inv;
    cmx = fmaxf(cmx, fmaxf(fmaxf(acc0[r], acc1[r]), fmaxf(acc2[r], acc3[r])));
  }
  cmx = fmaxf(cmx, xor16(cmx));  // other 32 channels live in partner lane
  float cs = 0.0f;
#pragma unroll
  for (int r = 0; r < 8; ++r) {
    acc0[r] = __expf(acc0[r] - cmx);
    acc1[r] = __expf(acc1[r] - cmx);
    acc2[r] = __expf(acc2[r] - cmx);
    acc3[r] = __expf(acc3[r] - cmx);
    cs += (acc0[r] + acc1[r]) + (acc2[r] + acc3[r]);
  }
  cs += xor16(cs);
  float cis = 1.0f / cs;

  // y^T accumulators are directly the B-fragments of out^T = out_w @ y^T
  v16h yb0, yb1;
#pragma unroll
  for (int r = 0; r < 8; ++r) {
    yb0[r]     = (_Float16)(acc0[r] * cis);  // K(channel) = hb*8 + r
    yb0[8 + r] = (_Float16)(acc1[r] * cis);  // K = 16 + hb*8 + r
    yb1[r]     = (_Float16)(acc2[r] * cis);  // K = 32 + hb*8 + r
    yb1[8 + r] = (_Float16)(acc3[r] * cis);  // K = 48 + hb*8 + r
  }

  // --- out^T = out_w @ y^T, + bias + residual, stored in NCHW layout ---
  const int token = qbase + lx;
#pragma unroll
  for (int jt = 0; jt < 8; ++jt) {
    v16h aw0 = load_frag16(owh, jt * 16 + lx, CI, 0, hb);
    v16h aw1 = load_frag16(owh, jt * 16 + lx, CI, 32, hb);
    v8f o = {};
    o = wmma_f16(aw0, yb0, o);
    o = wmma_f16(aw1, yb1, o);
#pragma unroll
    for (int r = 0; r < 8; ++r) {
      int c = jt * 16 + hb * 8 + r;          // output channel (uniform per half)
      long idx = (long)c * N_TOK + token;    // lanes 0-15: consecutive tokens
      out[idx] = o[r] + out_b[c] + x[idx];
    }
  }
}

extern "C" void kernel_launch(void* const* d_in, const int* in_sizes, int n_in,
                              void* d_out, int out_size, void* d_ws, size_t ws_size,
                              hipStream_t stream) {
  const float* x  = (const float*)d_in[0];
  const float* tw = (const float*)d_in[1];
  const float* tb = (const float*)d_in[2];
  const float* pw = (const float*)d_in[3];
  const float* pb = (const float*)d_in[4];
  const float* gw = (const float*)d_in[5];
  const float* gb = (const float*)d_in[6];
  const float* ow = (const float*)d_in[7];
  const float* ob = (const float*)d_in[8];
  float* out = (float*)d_out;

  // workspace: theta (NxCi), phi (NxCi), gT (CixN), out_w f16 — ~3.6 MB
  char* wsb = (char*)d_ws;
  _Float16* theta = (_Float16*)wsb;
  _Float16* phi   = (_Float16*)(wsb + (size_t)N_TOK * CI * 2);
  _Float16* gT    = (_Float16*)(wsb + (size_t)N_TOK * CI * 4);
  _Float16* owh   = (_Float16*)(wsb + (size_t)N_TOK * CI * 6);

  proj_kernel<<<N_TOK / 256, 256, 0, stream>>>(x, tw, tb, pw, pb, gw, gb, ow,
                                               theta, phi, gT, owh);
  flash_kernel<<<N_TOK / 128, 256, 0, stream>>>(theta, phi, gT, owh, ob, x, out);
}
